// SharpRNN_79370995630159
// MI455X (gfx1250) — matrix operand order repaired and checked by reading, not compiled, not run on hardware
//
#include <hip/hip_runtime.h>

typedef __attribute__((ext_vector_type(16))) _Float16 v16h;
typedef __attribute__((ext_vector_type(8)))  float    v8f;
typedef __attribute__((ext_vector_type(4)))  float    f32x4;

#define GRU_B 256
#define GRU_T 4096
#define GRU_C 16
#define GRU_H 20
#define GRU_G 60   // 3*H gate rows
#define GSTR  68   // padded gate stride (bank-conflict free: (4b+g)%64 distinct)
#define HSTR  36   // padded hidden stride (36b mod 64 distinct across b=0..15)

__device__ __forceinline__ float fast_sigmoid(float x) {
  return __builtin_amdgcn_rcpf(1.0f + __expf(-x));
}
__device__ __forceinline__ float fast_tanh(float x) {
  float e = __expf(2.0f * x);
  return 1.0f - 2.0f * __builtin_amdgcn_rcpf(e + 1.0f);
}

__device__ __forceinline__ v8f wmma16(v16h a, v16h b, v8f c) {
  // D = A(16x32 f16) * B(32x16 f16) + C(16x16 f32)
  return __builtin_amdgcn_wmma_f32_16x16x32_f16(false, a, false, b, (short)0, c,
                                                false, false);
}

__device__ __forceinline__ void store_d(float* p, v8f a) {
  f32x4 lo, hi;
  lo[0] = a[0]; lo[1] = a[1]; lo[2] = a[2]; lo[3] = a[3];
  hi[0] = a[4]; hi[1] = a[5]; hi[2] = a[6]; hi[3] = a[7];
  *(f32x4*)p       = lo;   // ds_store_b128
  *(f32x4*)(p + 4) = hi;
}

// B-matrix (32x16, f16): lane L holds column n=L%16; element e <-> k = e + 16*(L/16)
__device__ __forceinline__ v16h load_b16(const float* p) {
  f32x4 q0 = *(const f32x4*)(p);
  f32x4 q1 = *(const f32x4*)(p + 4);
  f32x4 q2 = *(const f32x4*)(p + 8);
  f32x4 q3 = *(const f32x4*)(p + 12);
  v16h r;
  #pragma unroll
  for (int e = 0; e < 4; ++e) {
    r[e]      = (_Float16)q0[e];
    r[e + 4]  = (_Float16)q1[e];
    r[e + 8]  = (_Float16)q2[e];
    r[e + 12] = (_Float16)q3[e];
  }
  return r;
}

// A-matrix (16x32, f16): lane (m=L%16, hh=L/16); element e <-> k = e + 16*hh
// W is [60, 16] row-major; K padded 16->32 with zeros.
__device__ __forceinline__ void load_wA_k16(const float* W, v16h* A, int m, int hh) {
  #pragma unroll
  for (int gt = 0; gt < 4; ++gt) {
    v16h a = {};
    int g = gt * 16 + m;
    if (g < GRU_G) {
      const float* row = W + g * 16;
      #pragma unroll
      for (int e = 0; e < 16; ++e) if (hh == 0) a[e] = (_Float16)row[e];   // k = e
    }
    A[gt] = a;
  }
}

// W is [60, 20] row-major; K padded 20->32 with zeros.
__device__ __forceinline__ void load_wA_k20(const float* W, v16h* A, int m, int hh) {
  #pragma unroll
  for (int gt = 0; gt < 4; ++gt) {
    v16h a = {};
    int g = gt * 16 + m;
    if (g < GRU_G) {
      const float* row = W + g * 20;
      if (hh == 0) {
        #pragma unroll
        for (int e = 0; e < 16; ++e) a[e] = (_Float16)row[e];       // k=0..15
      } else {
        #pragma unroll
        for (int e = 0; e < 4; ++e)  a[e] = (_Float16)row[e + 16];  // k=16..19
      }
    }
    A[gt] = a;
  }
}

__global__ __launch_bounds__(32, 1) void gru_fused_kernel(
    const float* __restrict__ x, const float* __restrict__ h0,
    const float* __restrict__ Wih0, const float* __restrict__ Whh0,
    const float* __restrict__ bih0, const float* __restrict__ bhh0,
    const float* __restrict__ Wih1, const float* __restrict__ Whh1,
    const float* __restrict__ bih1, const float* __restrict__ bhh1,
    const float* __restrict__ Wo, const float* __restrict__ bo,
    float* __restrict__ out)
{
  __shared__ __align__(16) float sG0[16 * GSTR];      // input-gate pre-acts [batch][gate]
  __shared__ __align__(16) float sG1[16 * GSTR];      // hidden-gate pre-acts
  __shared__ __align__(16) float sH[2][16 * HSTR];    // h state per layer, j<32 (j>=20 == 0)
  __shared__ float sBias[4][64];                      // b_ih0, b_hh0, b_ih1, b_hh1
  __shared__ float sWo[32];

  const int lane  = threadIdx.x;   // wave32
  const int m     = lane & 15;     // batch row within tile / A-matrix row
  const int hh    = lane >> 4;     // lane half
  const int bt    = blockIdx.x;    // batch tile 0..15
  const int bglob = bt * 16 + m;

  // ---- init LDS ----
  for (int i = lane; i < 2 * 16 * HSTR; i += 32) ((float*)sH)[i] = 0.0f;
  __syncthreads();
  for (int i = lane; i < 16 * 2 * GRU_H; i += 32) {
    int bl = i / (2 * GRU_H);
    int l  = (i / GRU_H) & 1;
    int j  = i % GRU_H;
    sH[l][bl * HSTR + j] = h0[((size_t)(bt * 16 + bl) * 2 + l) * GRU_H + j];
  }
  for (int i = lane; i < 4 * 64; i += 32) {
    int which = i >> 6, j = i & 63;
    const float* src = (which == 0) ? bih0 : (which == 1) ? bhh0
                      : (which == 2) ? bih1 : bhh1;
    sBias[which][j] = (j < GRU_G) ? src[j] : 0.0f;
  }
  if (lane < 32) sWo[lane] = (lane < GRU_H) ? Wo[lane] : 0.0f;
  const float bO = bo[0];

  // ---- weights resident in VGPRs as pre-swizzled A-matrices ----
  v16h Aih0[4], Ahh0[4], Aih1[4], Ahh1[4];
  load_wA_k16(Wih0, Aih0, m, hh);
  load_wA_k20(Whh0, Ahh0, m, hh);
  load_wA_k20(Wih1, Aih1, m, hh);
  load_wA_k20(Whh1, Ahh1, m, hh);

  __syncthreads();

  v16h h0B = load_b16(&sH[0][m * HSTR + 16 * hh]);
  v16h h1B = load_b16(&sH[1][m * HSTR + 16 * hh]);

  const float* xlane = x + (size_t)bglob * GRU_T * GRU_C;

  for (int t = 0; t < GRU_T; ++t) {
    // ---- x_t as B-matrix (K=16 channels x N=16 batch); lanes 16..31 pad k=16..31 -> 0
    v16h xB = {};
    if (hh == 0) {
      const float* px = xlane + (size_t)t * GRU_C;
      __builtin_prefetch(px + 8 * GRU_C, 0, 0);   // global_prefetch_b8 ~8 steps ahead
      f32x4 q0 = *(const f32x4*)(px);
      f32x4 q1 = *(const f32x4*)(px + 4);
      f32x4 q2 = *(const f32x4*)(px + 8);
      f32x4 q3 = *(const f32x4*)(px + 12);
      #pragma unroll
      for (int e = 0; e < 4; ++e) {
        xB[e]      = (_Float16)q0[e];
        xB[e + 4]  = (_Float16)q1[e];
        xB[e + 8]  = (_Float16)q2[e];
        xB[e + 12] = (_Float16)q3[e];
      }
    }

    // ---- layer 0: gate pre-activations, D = W * in (gate rows x batch cols) ----
    #pragma unroll
    for (int gt = 0; gt < 4; ++gt) {
      v8f z0 = {};
      v8f ax = wmma16(Aih0[gt], xB, z0);
      store_d(&sG0[m * GSTR + gt * 16 + hh * 8], ax);
      v8f z1 = {};
      v8f ah = wmma16(Ahh0[gt], h0B, z1);
      store_d(&sG1[m * GSTR + gt * 16 + hh * 8], ah);
    }
    __syncthreads();

    // ---- layer 0 pointwise: lane (m, hh) handles j = hh*10 .. hh*10+9 ----
    {
      const int gi = m * GSTR;
      #pragma unroll
      for (int jj = 0; jj < 10; ++jj) {
        const int j = hh * 10 + jj;
        float r = fast_sigmoid(sG0[gi + j] + sG1[gi + j] +
                               sBias[0][j] + sBias[1][j]);
        float z = fast_sigmoid(sG0[gi + 20 + j] + sG1[gi + 20 + j] +
                               sBias[0][20 + j] + sBias[1][20 + j]);
        float n = fast_tanh(sG0[gi + 40 + j] + sBias[0][40 + j] +
                            r * (sG1[gi + 40 + j] + sBias[1][40 + j]));
        float hp = sH[0][m * HSTR + j];
        sH[0][m * HSTR + j] = (1.0f - z) * n + z * hp;
      }
    }
    __syncthreads();

    h0B = load_b16(&sH[0][m * HSTR + 16 * hh]);   // h0_new feeds layer 1

    // ---- layer 1 GEMMs ----
    #pragma unroll
    for (int gt = 0; gt < 4; ++gt) {
      v8f z0 = {};
      v8f ax = wmma16(Aih1[gt], h0B, z0);
      store_d(&sG0[m * GSTR + gt * 16 + hh * 8], ax);
      v8f z1 = {};
      v8f ah = wmma16(Ahh1[gt], h1B, z1);
      store_d(&sG1[m * GSTR + gt * 16 + hh * 8], ah);
    }
    __syncthreads();

    // ---- layer 1 pointwise + 1-unit projection ----
    float part = 0.0f;
    {
      const int gi = m * GSTR;
      #pragma unroll
      for (int jj = 0; jj < 10; ++jj) {
        const int j = hh * 10 + jj;
        float r = fast_sigmoid(sG0[gi + j] + sG1[gi + j] +
                               sBias[2][j] + sBias[3][j]);
        float z = fast_sigmoid(sG0[gi + 20 + j] + sG1[gi + 20 + j] +
                               sBias[2][20 + j] + sBias[3][20 + j]);
        float n = fast_tanh(sG0[gi + 40 + j] + sBias[2][40 + j] +
                            r * (sG1[gi + 40 + j] + sBias[3][40 + j]));
        float hp = sH[1][m * HSTR + j];
        float hn = (1.0f - z) * n + z * hp;
        sH[1][m * HSTR + j] = hn;
        part += hn * sWo[j];
      }
    }
    __syncthreads();

    h1B = load_b16(&sH[1][m * HSTR + 16 * hh]);

    part += __shfl_xor(part, 16, 32);
    if (hh == 0) out[(size_t)bglob * GRU_T + t] = part + bO;
  }

  // ---- final hidden states: h_n [B, 2, H] appended after o [B, T, 1] ----
  __syncthreads();
  float* hn_out = out + (size_t)GRU_B * GRU_T;
  for (int i = lane; i < 16 * 2 * GRU_H; i += 32) {
    int bl = i / (2 * GRU_H);
    int l  = (i / GRU_H) & 1;
    int j  = i % GRU_H;
    hn_out[((size_t)(bt * 16 + bl) * 2 + l) * GRU_H + j] = sH[l][bl * HSTR + j];
  }
}

extern "C" void kernel_launch(void* const* d_in, const int* in_sizes, int n_in,
                              void* d_out, int out_size, void* d_ws, size_t ws_size,
                              hipStream_t stream) {
  (void)in_sizes; (void)n_in; (void)d_ws; (void)ws_size; (void)out_size;
  const float* x    = (const float*)d_in[0];
  const float* h0   = (const float*)d_in[1];
  const float* Wih0 = (const float*)d_in[2];
  const float* Whh0 = (const float*)d_in[3];
  const float* bih0 = (const float*)d_in[4];
  const float* bhh0 = (const float*)d_in[5];
  const float* Wih1 = (const float*)d_in[6];
  const float* Whh1 = (const float*)d_in[7];
  const float* bih1 = (const float*)d_in[8];
  const float* bhh1 = (const float*)d_in[9];
  const float* Wo   = (const float*)d_in[10];
  const float* bo   = (const float*)d_in[11];
  float* out = (float*)d_out;

  gru_fused_kernel<<<dim3(16), dim3(32), 0, stream>>>(
      x, h0, Wih0, Whh0, bih0, bhh0, Wih1, Whh1, bih1, bhh1, Wo, bo, out);
}
